// CoReGD_16312285790594
// MI455X (gfx1250) — matrix operand order repaired and checked by reading, compiled, not run on hardware
//
#include <hip/hip_runtime.h>

// CDNA5 / gfx1250: wave32, WMMA f16 -> f32 16x16x32.
typedef __attribute__((ext_vector_type(16))) _Float16 v16h;
typedef __attribute__((ext_vector_type(8)))  _Float16 v8h;
typedef __attribute__((ext_vector_type(8)))  float    v8f;

#define MT     64      // rows (edges/nodes) per workgroup
#define RT     4       // 16-row tiles per workgroup (MT/16)
#define NW     8       // waves per workgroup
#define BLK    256     // threads per workgroup
#define NMAXW  512     // max layer width
#define HIDF   128
#define KNN_K  8
#define LN_EPS 1e-5f

// ---------------------------------------------------------------------------
// Swizzled LDS activation layout == WMMA 16-bit A-matrix 16x32 fragment layout
// (ISA 7.12.2).  Key property: an 8-aligned K-chunk maps to 16 CONTIGUOUS
// bytes of one lane's fragment -> vector (v8h) stores/loads everywhere.
//   rt = row/16, m = row%16, f = k/32, kk = k%32
//   hi = (kk>>3)&1, j0 = ((kk>>4)<<3), lane = m + 16*hi
//   index = ((rt*16 + f)*32 + lane)*16 + j0 + (kk&7)
// ---------------------------------------------------------------------------
__device__ __forceinline__ int xsf_chunk_index(int row, int k) {  // k % 8 == 0
  int rt = row >> 4;
  int m  = row & 15;
  int f  = k >> 5;
  int kk = k & 31;
  int hi = (kk >> 3) & 1;
  int j0 = (kk >> 4) << 3;
  return ((rt * 16 + f) * 32 + (m + (hi << 4))) * 16 + j0;
}

// ---------------------------------------------------------------------------
// One GEMM layer: Ys[MT][Np] (f16) = Xs(frag layout, Kp wide) @ Wt^T + bias.
// Wt layout: [Np][Kp] f16 (pre-transposed) -> B fragment is two v8h loads.
// Waves partition N-tiles; each B fragment is reused across RT row-tiles
// (4 WMMAs per 1KB weight fetch) and loaded once per workgroup.
// ---------------------------------------------------------------------------
__device__ __forceinline__ void layer_compute(
    const _Float16* __restrict__ XsF, _Float16* __restrict__ Ys,
    const _Float16* __restrict__ Wt, const float* __restrict__ bias,
    int Kp, int Np, int n_act)
{
  const int tid  = threadIdx.x;
  const int w    = tid >> 5;       // wave id: owns nt = w, w+NW, ...
  const int lane = tid & 31;
  const int lm   = lane & 15;
  const int hi   = lane >> 4;
  const int nk   = Kp >> 5;
  const int nn   = Np >> 4;
  const _Float16* ap = XsF + lane * 16;   // + rt*8192 + f*512
  for (int nt = w; nt < nn; nt += NW) {
    const int col = nt * 16 + lm;
    union Acc { v8f v; float f[8]; } acc[RT];
#pragma unroll
    for (int rt = 0; rt < RT; ++rt)
#pragma unroll
      for (int r = 0; r < 8; ++r) acc[rt].f[r] = 0.0f;
    const _Float16* wp = Wt + (size_t)col * Kp + hi * 8;   // 16B aligned
    for (int f = 0; f < nk; ++f) {
      v8h blo = *(const v8h*)(wp + f * 32);
      v8h bhi = *(const v8h*)(wp + f * 32 + 16);
      v16h bv = __builtin_shufflevector(blo, bhi, 0, 1, 2, 3, 4, 5, 6, 7, 8, 9, 10, 11, 12, 13, 14, 15);  // global (B frag)
#pragma unroll
      for (int rt = 0; rt < RT; ++rt) {
        v16h av = *(const v16h*)(ap + rt * 8192 + f * 512); // ds_load (A frag)
        acc[rt].v = __builtin_amdgcn_wmma_f32_16x16x32_f16(
            false, av, false, bv, (short)0, acc[rt].v, false, false);
      }
    }
    const float b = (bias && col < n_act) ? bias[col] : 0.0f;
#pragma unroll
    for (int rt = 0; rt < RT; ++rt)
#pragma unroll
      for (int r = 0; r < 8; ++r) {                        // C layout: VGPR r ->
        int row = rt * 16 + (hi << 3) + r;                 // M = r + 8*hi
        Ys[row * NMAXW + col] = (_Float16)(acc[rt].f[r] + b);
      }
  }
}

// ---------------------------------------------------------------------------
// LayerNorm + ReLU over Ys rows; either re-stage into XsF (intermediate layer)
// or emit (final layer): 0=f16 store, 1=f32 store, 2=atomic scatter-add.
// nOut must be a multiple of 8 (true for 128/256/512).
// ---------------------------------------------------------------------------
__device__ __forceinline__ void post_ln(
    const _Float16* __restrict__ Ys, float* __restrict__ rmean,
    float* __restrict__ rrstd,
    const float* __restrict__ g, const float* __restrict__ bt, int nOut,
    _Float16* __restrict__ XsF_dst, int emit_mode,
    _Float16* __restrict__ out16, float* __restrict__ out32,
    float* __restrict__ aggout, const int* __restrict__ dst,
    int row0, int R, int d_out)
{
  const int tid = threadIdx.x;
  const int C8  = nOut >> 3;
  if (tid < MT) {
    float s = 0.0f, ss = 0.0f;
    const _Float16* yr = Ys + tid * NMAXW;
    for (int c8 = 0; c8 < C8; ++c8) {
      v8h y = *(const v8h*)(yr + (c8 << 3));
#pragma unroll
      for (int t = 0; t < 8; ++t) { float f = (float)y[t]; s += f; ss += f * f; }
    }
    float m = s / (float)nOut;
    float v = ss / (float)nOut - m * m;
    rmean[tid] = m;
    rrstd[tid] = rsqrtf(fmaxf(v, 0.0f) + LN_EPS);
  }
  __syncthreads();
  for (int i = tid; i < MT * C8; i += blockDim.x) {
    int row = i / C8, c8 = i - row * C8;
    int c0 = c8 << 3;
    v8h y = *(const v8h*)(Ys + row * NMAXW + c0);
    float mm = rmean[row], rs = rrstd[row];
    union { v8h h; _Float16 e[8]; } oh;
    float vf[8];
#pragma unroll
    for (int t = 0; t < 8; ++t) {
      float v = fmaxf(((float)y[t] - mm) * rs * g[c0 + t] + bt[c0 + t], 0.0f);
      vf[t] = v;
      oh.e[t] = (_Float16)v;
    }
    if (XsF_dst) *(v8h*)(XsF_dst + xsf_chunk_index(row, c0)) = oh.h;
    int gr = row0 + row;
    if (emit_mode >= 0 && gr < R) {
      if (emit_mode == 0) {
        *(v8h*)(out16 + (size_t)gr * d_out + c0) = oh.h;
      } else if (emit_mode == 1) {
#pragma unroll
        for (int t = 0; t < 8; ++t) out32[(size_t)gr * d_out + c0 + t] = vf[t];
      } else {
        float* ao = aggout + (size_t)dst[gr] * d_out + c0;
#pragma unroll
        for (int t = 0; t < 8; ++t) atomicAdd(ao + t, vf[t]);
      }
    }
  }
}

// ---------------------------------------------------------------------------
// Fused 3-layer MLP (Linear+LN+ReLU ×2, Linear[+LN+ReLU]) over MT rows/WG.
// gmode: 0 = f32 matrix, 1 = f16 matrix, 2 = edge gather [h[dst],h[src],ea?],
//        3 = residual h16 + agg (f32).
// LDS: XsF 64 KB + Ys(f16) 64 KB + stats -> ~128.5 KB => 2 WGs per WGP.
// ---------------------------------------------------------------------------
__global__ __launch_bounds__(BLK) void fused_mlp3_kernel(
    int R, int gmode,
    const float* __restrict__ Xf32, const _Float16* __restrict__ Xf16,
    const _Float16* __restrict__ Hg, const float* __restrict__ Agg,
    const int* __restrict__ src, const int* __restrict__ dst,
    const float* __restrict__ ea, int has_ea,
    int d0, int K0p, int d1, int d2, int d3p, int d3,
    const _Float16* __restrict__ Wt0, const float* __restrict__ b0,
    const float* __restrict__ g0, const float* __restrict__ bt0,
    const _Float16* __restrict__ Wt1, const float* __restrict__ b1,
    const float* __restrict__ g1, const float* __restrict__ bt1,
    const _Float16* __restrict__ Wt2, const float* __restrict__ b2,
    const float* __restrict__ g2, const float* __restrict__ bt2,
    int last_ln, int out_mode,
    _Float16* __restrict__ Out16, float* __restrict__ Out32,
    float* __restrict__ AggOut)
{
  extern __shared__ char smem[];
  _Float16* XsF   = (_Float16*)smem;                       // RT*16*32*16 h = 64 KB
  _Float16* Ys    = (_Float16*)(smem + 65536);             // 64*512 f16   = 64 KB
  float*    rmean = (float*)(smem + 65536 + 65536);        // 64 f32
  float*    rrstd = rmean + MT;                            // 64 f32

  const int tid  = threadIdx.x;
  const int row0 = blockIdx.x * MT;

  // ---- stage layer-0 input (zero-padded to K0p), 8-half vector chunks ----
  {
    const int C8 = K0p >> 3;
    for (int i = tid; i < MT * C8; i += blockDim.x) {
      int row = i / C8, c8 = i - row * C8;
      int k = c8 << 3;
      int gr = row0 + row;
      union { v8h h; _Float16 e[8]; } val;
#pragma unroll
      for (int t = 0; t < 8; ++t) val.e[t] = (_Float16)0.0f;
      if (gr < R) {
        if (gmode == 1) {
          if (k < d0) val.h = *(const v8h*)(Xf16 + (size_t)gr * d0 + k);
        } else if (gmode == 2) {
          if (k < HIDF)
            val.h = *(const v8h*)(Hg + (size_t)dst[gr] * HIDF + k);
          else if (k < 2 * HIDF)
            val.h = *(const v8h*)(Hg + (size_t)src[gr] * HIDF + (k - HIDF));
          else if (has_ea && k == 2 * HIDF)
            val.e[0] = (_Float16)ea[gr];
        } else if (gmode == 3) {
          if (k < d0) {
            v8h hv = *(const v8h*)(Hg + (size_t)gr * HIDF + k);
            v8f av = *(const v8f*)(Agg + (size_t)gr * HIDF + k);
#pragma unroll
            for (int t = 0; t < 8; ++t) val.e[t] = (_Float16)((float)hv[t] + av[t]);
          }
        } else {  // gmode 0: f32 input, tiny d0 -> elementwise
#pragma unroll
          for (int t = 0; t < 8; ++t) {
            int kk = k + t;
            if (kk < d0) val.e[t] = (_Float16)Xf32[(size_t)gr * d0 + kk];
          }
        }
      }
      *(v8h*)(XsF + xsf_chunk_index(row, k)) = val.h;
    }
  }
  __syncthreads();

  // ---- layer 0 ----
  layer_compute(XsF, Ys, Wt0, b0, K0p, d1, d1);
  __syncthreads();
  post_ln(Ys, rmean, rrstd, g0, bt0, d1, XsF, -1,
          nullptr, nullptr, nullptr, nullptr, row0, R, d1);
  __syncthreads();

  // ---- layer 1 ----
  layer_compute(XsF, Ys, Wt1, b1, d1, d2, d2);
  __syncthreads();
  post_ln(Ys, rmean, rrstd, g1, bt1, d2, XsF, -1,
          nullptr, nullptr, nullptr, nullptr, row0, R, d2);
  __syncthreads();

  // ---- layer 2 ----
  layer_compute(XsF, Ys, Wt2, b2, d2, d3p, d3);
  __syncthreads();
  if (last_ln) {
    post_ln(Ys, rmean, rrstd, g2, bt2, d3, nullptr, out_mode,
            Out16, Out32, AggOut, dst, row0, R, d3);
  } else {
    for (int i = tid; i < MT * d3; i += blockDim.x) {
      int row = i / d3, c = i - row * d3;
      int gr = row0 + row;
      if (gr < R) {
        float y = (float)Ys[row * NMAXW + c];
        if (out_mode == 1)      Out32[(size_t)gr * d3 + c] = y;
        else if (out_mode == 0) Out16[(size_t)gr * d3 + c] = (_Float16)y;
      }
    }
  }
}

// ---------------------------------------------------------------------------
// Weight prep: W[K][N] f32 -> Wt[Np][Kp] f16 (transpose + pad with zeros).
// ---------------------------------------------------------------------------
__global__ void prep_weight_kernel(const float* __restrict__ W,
                                   _Float16* __restrict__ Wt,
                                   int K, int N, int Kp, int Np) {
  int i = blockIdx.x * blockDim.x + threadIdx.x;
  if (i >= Kp * Np) return;
  int n = i / Kp, k = i - n * Kp;
  float v = (k < K && n < N) ? W[(size_t)k * N + n] : 0.0f;
  Wt[i] = (_Float16)v;
}

__global__ void zero_kernel(float* __restrict__ p, int n) {
  int i = blockIdx.x * blockDim.x + threadIdx.x;
  if (i < n) p[i] = 0.0f;
}

// ---------------------------------------------------------------------------
// Brute-force 2-D kNN (k=8): one 256-thread block per query node.
// ---------------------------------------------------------------------------
__global__ __launch_bounds__(256) void knn_kernel(
    const float* __restrict__ pos, int N,
    int* __restrict__ ksrc, int* __restrict__ kdst)
{
  __shared__ float sd[256 * KNN_K];
  __shared__ int   si[256 * KNN_K];
  const int q   = blockIdx.x;
  const int tid = threadIdx.x;
  const float qx = pos[2 * q], qy = pos[2 * q + 1];
  float bd[KNN_K]; int bi[KNN_K];
#pragma unroll
  for (int t = 0; t < KNN_K; ++t) { bd[t] = 3.4e38f; bi[t] = 0; }
  for (int j = tid; j < N; j += 256) {
    if (j == q) continue;
    float dx = pos[2 * j] - qx, dy = pos[2 * j + 1] - qy;
    float d = dx * dx + dy * dy;
    if (d < bd[KNN_K - 1]) {
      int p = KNN_K - 1;
      while (p > 0 && bd[p - 1] > d) { bd[p] = bd[p - 1]; bi[p] = bi[p - 1]; --p; }
      bd[p] = d; bi[p] = j;
    }
  }
#pragma unroll
  for (int t = 0; t < KNN_K; ++t) { sd[tid * KNN_K + t] = bd[t]; si[tid * KNN_K + t] = bi[t]; }
  for (int off = 1; off < 256; off <<= 1) {
    __syncthreads();
    if ((tid & (2 * off - 1)) == 0) {
      float md[KNN_K]; int mi[KNN_K];
      int a = tid * KNN_K, b = (tid + off) * KNN_K;
      int ia = 0, ib = 0;
#pragma unroll
      for (int t = 0; t < KNN_K; ++t) {
        if (sd[a + ia] <= sd[b + ib]) { md[t] = sd[a + ia]; mi[t] = si[a + ia]; ++ia; }
        else                          { md[t] = sd[b + ib]; mi[t] = si[b + ib]; ++ib; }
      }
#pragma unroll
      for (int t = 0; t < KNN_K; ++t) { sd[a + t] = md[t]; si[a + t] = mi[t]; }
    }
  }
  __syncthreads();
  if (tid < KNN_K) {
    ksrc[q * KNN_K + tid] = si[tid];
    kdst[q * KNN_K + tid] = q;
  }
}

// ---------------------------------------------------------------------------
// Host orchestration
// ---------------------------------------------------------------------------
extern "C" void kernel_launch(void* const* d_in, const int* in_sizes, int n_in,
                              void* d_out, int out_size, void* d_ws, size_t ws_size,
                              hipStream_t stream) {
  (void)in_sizes; (void)n_in; (void)out_size; (void)ws_size;
  const int N = 8192, E = 262144, IN_CH = 8, OUT_CH = 2;

  // ---- input mapping: dict insertion-order flatten of setup_inputs() ----
  const float* x         = (const float*)d_in[0];
  const float* edge_attr = (const float*)d_in[1];
  int idx = 2;
  struct MLPp { const float *W[3], *b[3], *g[3], *bt[3]; };
  auto grab = [&](int ng) {
    MLPp m{};
    for (int i = 0; i < 3; ++i)  m.W[i]  = (const float*)d_in[idx++];
    for (int i = 0; i < 3; ++i)  m.b[i]  = (const float*)d_in[idx++];
    for (int i = 0; i < ng; ++i) m.g[i]  = (const float*)d_in[idx++];
    for (int i = 0; i < ng; ++i) m.bt[i] = (const float*)d_in[idx++];
    return m;
  };
  MLPp enc       = grab(3);
  MLPp conv_node = grab(3);
  MLPp conv_edge = grab(3);
  MLPp alt_node  = grab(3);
  MLPp alt_edge  = grab(3);
  MLPp dec       = grab(2);                  // decoder: no LN on last layer
  const int* edge_index = (const int*)d_in[idx++];
  const int* src = edge_index;
  const int* dst = edge_index + E;
  // d_in[idx] = iterations (fixed 2 in this config)

  // ---- workspace carve ----
  char* ws = (char*)d_ws;
  size_t off = 0;
  auto carve = [&](size_t bytes) -> void* {
    void* p = ws + off;
    off = (off + bytes + 255) & ~(size_t)255;
    return p;
  };
  _Float16* hA  = (_Float16*)carve((size_t)N * HIDF * 2);
  _Float16* hB  = (_Float16*)carve((size_t)N * HIDF * 2);
  float*    agg = (float*)carve((size_t)N * HIDF * 4);
  float*    pos = (float*)carve((size_t)N * 2 * 4);
  int*     ksrc = (int*)carve((size_t)N * KNN_K * 4);
  int*     kdst = (int*)carve((size_t)N * KNN_K * 4);

  // ---- weight prep (f32 [K][N] -> f16 [Np][Kp], every call) ----
  struct WT { _Float16* w[3]; };
  auto prep = [&](const MLPp& m, const int* K, const int* Nd,
                  const int* Kp, const int* Np) {
    WT t{};
    for (int i = 0; i < 3; ++i) {
      int total = Kp[i] * Np[i];
      t.w[i] = (_Float16*)carve((size_t)total * 2);
      prep_weight_kernel<<<dim3((total + 255) / 256), dim3(256), 0, stream>>>(
          m.W[i], t.w[i], K[i], Nd[i], Kp[i], Np[i]);
    }
    return t;
  };
  const int encK[3]={IN_CH,256,256},   encN[3]={256,256,HIDF},  encKp[3]={32,256,256},  encNp[3]={256,256,128};
  const int nodK[3]={HIDF,256,256},    nodN[3]={256,256,HIDF},  nodKp[3]={128,256,256}, nodNp[3]={256,256,128};
  const int ceK[3]={2*HIDF+1,512,512}, ceN[3]={512,512,HIDF},   ceKp[3]={288,512,512},  ceNp[3]={512,512,128};
  const int aeK[3]={2*HIDF,512,512},   aeN[3]={512,512,HIDF},   aeKp[3]={256,512,512},  aeNp[3]={512,512,128};
  const int deK[3]={HIDF,256,256},     deN[3]={256,256,OUT_CH}, deKp[3]={128,256,256},  deNp[3]={256,256,16};
  WT wenc = prep(enc, encK, encN, encKp, encNp);
  WT wcn  = prep(conv_node, nodK, nodN, nodKp, nodNp);
  WT wce  = prep(conv_edge, ceK, ceN, ceKp, ceNp);
  WT wan  = prep(alt_node, nodK, nodN, nodKp, nodNp);
  WT wae  = prep(alt_edge, aeK, aeN, aeKp, aeNp);
  WT wdec = prep(dec, deK, deN, deKp, deNp);

  const size_t SMEM = 65536 + 65536 + 2 * MT * sizeof(float);  // ~128.5 KB
  const dim3 blk(BLK);
  auto gr = [](int R) { return dim3((R + MT - 1) / MT); };

  // ---- encoder: x -> hA ----
  fused_mlp3_kernel<<<gr(N), blk, SMEM, stream>>>(
      N, 0, x, nullptr, nullptr, nullptr, nullptr, nullptr, nullptr, 0,
      IN_CH, 32, 256, 256, 128, 128,
      wenc.w[0], enc.b[0], enc.g[0], enc.bt[0],
      wenc.w[1], enc.b[1], enc.g[1], enc.bt[1],
      wenc.w[2], enc.b[2], enc.g[2], enc.bt[2],
      1, 0, hA, nullptr, nullptr);

  // ---- one GIN step: edge MLP (scatter-add) + node MLP ----
  auto run_gin = [&](const WT& we, const MLPp& me, const WT& wn, const MLPp& mn,
                     const _Float16* hin, _Float16* hout,
                     int nE, const int* s, const int* d, const float* eatt,
                     int d0, int K0p) {
    zero_kernel<<<dim3((N * HIDF + 255) / 256), dim3(256), 0, stream>>>(agg, N * HIDF);
    fused_mlp3_kernel<<<gr(nE), blk, SMEM, stream>>>(
        nE, 2, nullptr, nullptr, hin, nullptr, s, d, eatt, eatt ? 1 : 0,
        d0, K0p, 512, 512, 128, 128,
        we.w[0], me.b[0], me.g[0], me.bt[0],
        we.w[1], me.b[1], me.g[1], me.bt[1],
        we.w[2], me.b[2], me.g[2], me.bt[2],
        1, 2, nullptr, nullptr, agg);
    fused_mlp3_kernel<<<gr(N), blk, SMEM, stream>>>(
        N, 3, nullptr, nullptr, hin, agg, nullptr, nullptr, nullptr, 0,
        HIDF, 128, 256, 256, 128, 128,
        wn.w[0], mn.b[0], mn.g[0], mn.bt[0],
        wn.w[1], mn.b[1], mn.g[1], mn.bt[1],
        wn.w[2], mn.b[2], mn.g[2], mn.bt[2],
        1, 0, hout, nullptr, nullptr);
  };
  auto run_dec = [&](const _Float16* hin, float* outp) {
    fused_mlp3_kernel<<<gr(N), blk, SMEM, stream>>>(
        N, 1, nullptr, hin, nullptr, nullptr, nullptr, nullptr, nullptr, 0,
        HIDF, 128, 256, 256, 16, OUT_CH,
        wdec.w[0], dec.b[0], dec.g[0], dec.bt[0],
        wdec.w[1], dec.b[1], dec.g[1], dec.bt[1],
        wdec.w[2], dec.b[2], nullptr, nullptr,
        0, 1, nullptr, outp, nullptr);
  };

  // ---- iteration body (ITERS-1 = 1) ----
  run_gin(wce, conv_edge, wcn, conv_node, hA, hB, E, src, dst, edge_attr, 2 * HIDF + 1, 288);
  run_dec(hB, pos);
  knn_kernel<<<dim3(N), dim3(256), 0, stream>>>(pos, N, ksrc, kdst);
  run_gin(wae, alt_edge, wan, alt_node, hB, hA, N * KNN_K, ksrc, kdst, nullptr, 2 * HIDF, 256);

  // ---- final conv + decoder -> d_out [N, 2] f32 ----
  run_gin(wce, conv_edge, wcn, conv_node, hA, hB, E, src, dst, edge_attr, 2 * HIDF + 1, 288);
  run_dec(hB, (float*)d_out);
}
